// MemorySentenceLookup_42185168781654
// MI455X (gfx1250) — matrix operand order repaired and checked by reading, not compiled, not run on hardware
//
#include <hip/hip_runtime.h>
#include <math.h>

typedef __attribute__((ext_vector_type(16))) _Float16 v16h;
typedef __attribute__((ext_vector_type(8)))  _Float16 v8h;
typedef __attribute__((ext_vector_type(4)))  _Float16 v4h;
typedef __attribute__((ext_vector_type(8)))  float    v8f;
typedef __attribute__((ext_vector_type(4)))  float    v4f;

#define B_       64
#define M_       512
#define D_       768
#define H1_      2048
#define H2_      1024
#define ROWS     32
#define NTHREADS 256

#if defined(__has_builtin)
#  if __has_builtin(__builtin_amdgcn_tanhf)
#    define TANHF(x) __builtin_amdgcn_tanhf(x)
#  else
#    define TANHF(x) tanhf(x)
#  endif
#else
#  define TANHF(x) tanhf(x)
#endif

__device__ __forceinline__ v8f wmma_f16(v16h a, v16h b, v8f c) {
  // D = A(16x32 f16) * B(32x16 f16) + C(16x16 f32)
  return __builtin_amdgcn_wmma_f32_16x16x32_f16(
      false, a, false, b, (short)0, c, false, false);
}

// ---- fp32 -> f16 weight conversion into workspace (stays L2-resident) ----
__global__ void cvt_f32_f16_kernel(const float* __restrict__ src,
                                   _Float16* __restrict__ dst, int n) {
  int i = blockIdx.x * blockDim.x + threadIdx.x;
  if (i < n) dst[i] = (_Float16)src[i];
}

// ---- qc[b][j] = b1[j] + query[b] @ W1[768:1536, j]  (shared by all 512 rows of b) ----
__global__ void qcontrib_kernel(const float* __restrict__ query,  // [64][768]
                                const float* __restrict__ W1,     // [1536][2048]
                                const float* __restrict__ b1,     // [2048]
                                float* __restrict__ qc) {         // [64][2048]
  const int j = blockIdx.x * blockDim.x + threadIdx.x;  // 0..2047
  const int b = blockIdx.y;                             // 0..63
  const float* qp = query + (size_t)b * D_;
  const float* wp = W1 + (size_t)D_ * H1_ + j;          // bottom half of W1
  float acc = b1[j];
  #pragma unroll 4
  for (int k = 0; k < D_; ++k) acc = fmaf(qp[k], wp[(size_t)k * H1_], acc);
  qc[(size_t)b * H1_ + j] = acc;
}

// ---- fused MLP over a 32-row tile; activations never leave LDS ----
__global__ __launch_bounds__(NTHREADS, 1)
void fused_mlp_kernel(const float* __restrict__ memory,   // [64][512][768]
                      const _Float16* __restrict__ W1h,   // [768][2048] f16 (top half)
                      const float* __restrict__ qc,       // [64][2048] (q@W1bot + b1)
                      const _Float16* __restrict__ W2h,   // [2048][1024] f16
                      const float* __restrict__ b2,       // [1024]
                      const float* __restrict__ W3,       // [1024]
                      const float* __restrict__ b3,       // [1]
                      float* __restrict__ out)            // [32768]
{
  extern __shared__ char smem[];
  _Float16* mems = (_Float16*)smem;                                    // [32][768]  48 KB
  _Float16* h1s  = (_Float16*)(smem + ROWS * D_ * 2);                  // [32][2048] 128 KB
  _Float16* h2s  = (_Float16*)(smem + ROWS * D_ * 2 + ROWS * H1_ * 2); // [32][1024] 64 KB

  const int wg   = blockIdx.x;
  const int row0 = wg * ROWS;          // global row base (b*512 + m0)
  const int b    = row0 / M_;
  const int m0   = row0 % M_;
  const int tid  = threadIdx.x;

  // ---------- stage memory tile (f32 -> f16) ----------
  {
    const int nv = ROWS * (D_ / 4);
    const float* mem_base = memory + ((size_t)b * M_ + m0) * D_;
    for (int i = tid; i < nv; i += NTHREADS) {
      int r = i / (D_ / 4), c4 = i % (D_ / 4);
      v4f v = ((const v4f*)(mem_base + (size_t)r * D_))[c4];
      ((v4h*)(mems + (size_t)r * D_))[c4] = __builtin_convertvector(v, v4h);
    }
  }
  __syncthreads();

  const int lane = tid & 31;
  const int wave = tid >> 5;
  const int hsel = lane >> 4;   // lane half-select per ISA 16-bit A layout
  const int lrow = lane & 15;
  const int rb   = wave & 1;    // row block (2 x 16 rows)
  const int cg   = wave >> 1;   // col group (4 groups x 4 tiles each pass)

  union { v16h v; v8h h[2]; } af;

  // ---------- layer 1: h1 = tanh(mem @ W1top + qc[b]) ----------
  {
    const _Float16* arow = mems + (size_t)(rb * 16 + lrow) * D_;
    for (int ct = cg * 4; ct < H1_ / 16; ct += 16) {
      const int col = ct * 16;                       // 4 tiles: col .. col+63
      v8f acc0 = {}, acc1 = {}, acc2 = {}, acc3 = {};
      const _Float16* w0 = W1h + col + lrow;
      __builtin_prefetch(w0 + 256, 0, 1);            // next tile-quad panel
      for (int k0 = 0; k0 < D_; k0 += 32) {
        af.h[0] = *(const v8h*)(arow + k0 + hsel * 8);       // K 0..7  / 8..15
        af.h[1] = *(const v8h*)(arow + k0 + 16 + hsel * 8);  // K 16..23/24..31
        const _Float16* wk = w0 + (size_t)k0 * H1_;
        v16h bf0, bf1, bf2, bf3;
        #pragma unroll
        for (int s = 0; s < 16; s++) {
          const size_t kr = (size_t)(((s & 8) << 1) + hsel * 8 + (s & 7)) * H1_;
          bf0[s] = wk[kr]; bf1[s] = wk[kr + 16]; bf2[s] = wk[kr + 32]; bf3[s] = wk[kr + 48];
        }
        acc0 = wmma_f16(af.v, bf0, acc0);   // 4 independent accumulator chains
        acc1 = wmma_f16(af.v, bf1, acc1);
        acc2 = wmma_f16(af.v, bf2, acc2);
        acc3 = wmma_f16(af.v, bf3, acc3);
      }
      const float* qcp = qc + (size_t)b * H1_ + col + lrow;
      const float q0 = qcp[0], q1 = qcp[16], q2 = qcp[32], q3 = qcp[48];
      const int mrow = rb * 16 + hsel * 8;
      _Float16* hp = h1s + (size_t)mrow * H1_ + col + lrow;
      #pragma unroll
      for (int v = 0; v < 8; v++) {
        hp[(size_t)v * H1_ + 0]  = (_Float16)TANHF(acc0[v] + q0);
        hp[(size_t)v * H1_ + 16] = (_Float16)TANHF(acc1[v] + q1);
        hp[(size_t)v * H1_ + 32] = (_Float16)TANHF(acc2[v] + q2);
        hp[(size_t)v * H1_ + 48] = (_Float16)TANHF(acc3[v] + q3);
      }
    }
  }
  __syncthreads();

  // ---------- layer 2: h2 = tanh(h1 @ W2 + b2) ----------
  {
    const _Float16* arow = h1s + (size_t)(rb * 16 + lrow) * H1_;
    for (int ct = cg * 4; ct < H2_ / 16; ct += 16) {
      const int col = ct * 16;
      v8f acc0 = {}, acc1 = {}, acc2 = {}, acc3 = {};
      const _Float16* w0 = W2h + col + lrow;
      __builtin_prefetch(w0 + 256, 0, 1);
      for (int k0 = 0; k0 < H1_; k0 += 32) {
        af.h[0] = *(const v8h*)(arow + k0 + hsel * 8);
        af.h[1] = *(const v8h*)(arow + k0 + 16 + hsel * 8);
        const _Float16* wk = w0 + (size_t)k0 * H2_;
        v16h bf0, bf1, bf2, bf3;
        #pragma unroll
        for (int s = 0; s < 16; s++) {
          const size_t kr = (size_t)(((s & 8) << 1) + hsel * 8 + (s & 7)) * H2_;
          bf0[s] = wk[kr]; bf1[s] = wk[kr + 16]; bf2[s] = wk[kr + 32]; bf3[s] = wk[kr + 48];
        }
        acc0 = wmma_f16(af.v, bf0, acc0);
        acc1 = wmma_f16(af.v, bf1, acc1);
        acc2 = wmma_f16(af.v, bf2, acc2);
        acc3 = wmma_f16(af.v, bf3, acc3);
      }
      const float q0 = b2[col + lrow],      q1 = b2[col + 16 + lrow];
      const float q2 = b2[col + 32 + lrow], q3 = b2[col + 48 + lrow];
      const int mrow = rb * 16 + hsel * 8;
      _Float16* hp = h2s + (size_t)mrow * H2_ + col + lrow;
      #pragma unroll
      for (int v = 0; v < 8; v++) {
        hp[(size_t)v * H2_ + 0]  = (_Float16)TANHF(acc0[v] + q0);
        hp[(size_t)v * H2_ + 16] = (_Float16)TANHF(acc1[v] + q1);
        hp[(size_t)v * H2_ + 32] = (_Float16)TANHF(acc2[v] + q2);
        hp[(size_t)v * H2_ + 48] = (_Float16)TANHF(acc3[v] + q3);
      }
    }
  }
  __syncthreads();

  // ---------- layer 3: out = h2 @ W3 + b3 (matvec, VALU) ----------
  {
    float* red = (float*)h1s;           // reuse dead h1 space for partials
    const int r = tid & 31, seg = tid >> 5;
    const _Float16* hrow = h2s + (size_t)r * H2_ + seg * 128;
    const float* w3 = W3 + seg * 128;
    float s = 0.f;
    #pragma unroll 8
    for (int c = 0; c < 128; c++) s += (float)hrow[c] * w3[c];
    red[seg * 32 + r] = s;
    __syncthreads();
    if (tid < 32) {
      float acc2 = b3[0];
      #pragma unroll
      for (int k = 0; k < 8; k++) acc2 += red[k * 32 + tid];
      out[row0 + tid] = acc2;
    }
  }
}

extern "C" void kernel_launch(void* const* d_in, const int* in_sizes, int n_in,
                              void* d_out, int out_size, void* d_ws, size_t ws_size,
                              hipStream_t stream) {
  (void)in_sizes; (void)n_in; (void)out_size; (void)ws_size;
  const float* query  = (const float*)d_in[0];
  const float* memory = (const float*)d_in[1];
  const float* W1     = (const float*)d_in[2];
  const float* b1     = (const float*)d_in[3];
  const float* W2     = (const float*)d_in[4];
  const float* b2     = (const float*)d_in[5];
  const float* W3     = (const float*)d_in[6];
  const float* b3     = (const float*)d_in[7];
  float* out = (float*)d_out;

  // workspace layout
  _Float16* W1h = (_Float16*)d_ws;                        // 768*2048 f16 = 3.0 MB
  _Float16* W2h = W1h + (size_t)D_ * H1_;                 // 2048*1024 f16 = 4.0 MB
  float*    qc  = (float*)(W2h + (size_t)H1_ * H2_);      // 64*2048 f32 = 0.5 MB

  const int n1 = D_ * H1_;        // top half of W1 only
  const int n2 = H1_ * H2_;
  cvt_f32_f16_kernel<<<(n1 + 255) / 256, 256, 0, stream>>>(W1, W1h, n1);
  cvt_f32_f16_kernel<<<(n2 + 255) / 256, 256, 0, stream>>>(W2, W2h, n2);
  qcontrib_kernel<<<dim3(H1_ / 256, B_), 256, 0, stream>>>(query, W1, b1, qc);

  const int nblocks = (B_ * M_) / ROWS;                   // 1024 WGs
  const size_t smem_bytes = (size_t)ROWS * D_ * 2         // mem tile: 48 KB
                          + (size_t)ROWS * H1_ * 2        // h1: 128 KB
                          + (size_t)ROWS * H2_ * 2;       // h2: 64 KB  => 240 KB
  (void)hipFuncSetAttribute((const void*)fused_mlp_kernel,
                            hipFuncAttributeMaxDynamicSharedMemorySize,
                            (int)smem_bytes);
  fused_mlp_kernel<<<nblocks, NTHREADS, smem_bytes, stream>>>(
      memory, W1h, qc, W2h, b2, W3, b3, out);
}
